// BilateralFilter_506806141157
// MI455X (gfx1250) — matrix operand order repaired and checked by reading, not compiled, not run on hardware
//
#include <hip/hip_runtime.h>

typedef __attribute__((ext_vector_type(16))) _Float16 v16h;
typedef __attribute__((ext_vector_type(8)))  _Float16 v8h;
typedef __attribute__((ext_vector_type(8)))  float    v8f;
typedef __attribute__((ext_vector_type(4)))  float    v4f;
typedef __attribute__((ext_vector_type(4)))  int      v4i;

#define KW     7
#define PADR   3
#define HH     256
#define WW     256
#define CIN    32
#define CG     3
#define TPX    16                  // pixels per wave tile (width)
#define TROWS  4                   // output rows per workgroup (one per wave)
#define UW     22                  // valid union width  (TPX + KW - 1)
#define UWP    24                  // padded union row stride (16B-aligned rows in f16)
#define UH     10                  // valid union height (TROWS + KW - 1)
#define UHP    11                  // padded height so j<192 stays in-array
#define KTOT   (KCH * 32)          // 192 padded K
#define KCH    6                   // K chunks of 32 (covers 7*24=168, padded to 192)
#define WSTR   52                  // Wl tap stride (pad 49 -> 52, 16B-aligned rows)

__global__ __launch_bounds__(128) void bilateral_wmma_kernel(
    const float* __restrict__ x,     // (2,32,256,256)
    const float* __restrict__ g,     // (2,3,256,256)
    const float* __restrict__ sig,   // scalar
    float* __restrict__ out)         // (2,32,256,256)
{
  // input tile, union layout: xS[c*UHP*UWP + r*UWP + col]; addr linear in j=dy*UWP+u
  __shared__ __align__(16) _Float16 xS[CIN * UHP * UWP];        // 16896 B
  __shared__ float gS[CG][UH][UW];                              //  2640 B
  __shared__ float Wl[TROWS][TPX][WSTR];                        // 13312 B
  __shared__ float invl[TROWS][TPX];                            //   256 B
  // normalized weights scattered onto padded union support (zero elsewhere)
  __shared__ __align__(16) _Float16 Bu[TROWS * TPX * KTOT];     // 24576 B

  const int tid = threadIdx.x;
  const int b   = blockIdx.z;
  const int h0  = blockIdx.y * TROWS;
  const int w0  = blockIdx.x * TPX;

  // ---- zero Bu (int4 stores: 24576B / 16 = 1536)
  for (int i = tid; i < (TROWS * TPX * KTOT) / 8; i += 128)
    ((v4i*)Bu)[i] = (v4i){0, 0, 0, 0};

  // ---- stage guide tile (f32, zero-padded like unfold)
  for (int i = tid; i < CG * UH * UW; i += 128) {
    int c = i / (UH * UW), rem = i % (UH * UW);
    int r = rem / UW, col = rem % UW;
    int gr = h0 - PADR + r, gc = w0 - PADR + col;
    float v = 0.f;
    if ((unsigned)gr < HH && (unsigned)gc < WW)
      v = g[((b * CG + c) * HH + gr) * WW + gc];
    gS[c][r][col] = v;
  }

  // ---- stage input tile as f16 into padded union layout (pad cells only ever
  //      meet B==0, any finite value is fine; zero when off-image)
  for (int i = tid; i < CIN * UHP * UWP; i += 128) {
    int c = i / (UHP * UWP), rem = i % (UHP * UWP);
    int r = rem / UWP, col = rem % UWP;
    int gr = h0 - PADR + r, gc = w0 - PADR + col;
    float v = 0.f;
    if ((unsigned)gr < HH && (unsigned)gc < WW)
      v = x[((b * CIN + c) * HH + gr) * WW + gc];
    xS[i] = (_Float16)v;
  }
  __syncthreads();

  const int wid  = tid >> 5;        // wave id -> output row h0+wid
  const int lane = tid & 31;
  const int p    = lane & 15;       // pixel within group / A row (channel index)
  const int hi   = lane >> 4;       // lane half
  const float sg = sig[0];
  const float inv_s2 = 1.f / (sg * sg);

  // ---- per-pixel weights: w = exp(-0.5*((ry^2+rx^2)/sigma^2 + sum_c diff^2))
  // lane handles pixel p; lane-halves split kx by parity (no divisions)
  {
    float cen0 = gS[0][wid + PADR][p + PADR];
    float cen1 = gS[1][wid + PADR][p + PADR];
    float cen2 = gS[2][wid + PADR][p + PADR];
#pragma unroll
    for (int dy = 0; dy < KW; ++dy) {
#pragma unroll
      for (int k2 = 0; k2 < 4; ++k2) {
        int kx = 2 * k2 + hi;
        if (kx < KW) {
          float d0 = gS[0][wid + dy][p + kx] - cen0;
          float d1 = gS[1][wid + dy][p + kx] - cen1;
          float d2 = gS[2][wid + dy][p + kx] - cen2;
          float d  = d0 * d0 + d1 * d1 + d2 * d2;
          float ry = (float)(dy - PADR), rx = (float)(kx - PADR);
          Wl[wid][p][dy * KW + kx] =
              __expf(-0.5f * ((ry * ry + rx * rx) * inv_s2 + d));
        }
      }
    }
  }
  __syncthreads();

  // ---- per-pixel normalization (exp>0 so sum>0; norm==0 guard moot)
  if (lane < TPX) {
    const v4f* wp = (const v4f*)&Wl[wid][lane][0];   // row base 208B-aligned
    float s = 0.f;
#pragma unroll
    for (int t4 = 0; t4 < 12; ++t4) {
      v4f v = wp[t4];
      s += v[0] + v[1] + v[2] + v[3];
    }
    s += Wl[wid][lane][48];
    invl[wid][lane] = 1.f / s;
  }
  __syncthreads();

  // ---- scatter normalized weights to union layout: j = dy*UWP + p + kx
  {
    const float invp = invl[wid][p];
    _Float16* bup = &Bu[(wid * TPX + p) * KTOT];
#pragma unroll
    for (int dy = 0; dy < KW; ++dy) {
#pragma unroll
      for (int k2 = 0; k2 < 4; ++k2) {
        int kx = 2 * k2 + hi;
        if (kx < KW)
          bup[dy * UWP + p + kx] = (_Float16)(Wl[wid][p][dy * KW + kx] * invp);
      }
    }
  }
  __syncthreads();

  // ---- B fragments: pure aligned v8h loads, address linear in j
  // B layout (32x16 f16): lane L: N=L%16, base=(L/16)*8; e<8 -> j=q*32+base+e ; e>=8 -> j=q*32+16+base+(e-8)
  const v8h* bp = (const v8h*)&Bu[(wid * TPX + p) * KTOT];
  v16h Bf[KCH];
#pragma unroll
  for (int q = 0; q < KCH; ++q) {
    v8h blo = bp[q * 4 + hi];
    v8h bhi = bp[q * 4 + 2 + hi];
    Bf[q] = __builtin_shufflevector(blo, bhi, 0, 1, 2, 3, 4, 5, 6, 7,
                                    8, 9, 10, 11, 12, 13, 14, 15);
  }

  const int h = h0 + wid;
#pragma unroll
  for (int half = 0; half < 2; ++half) {
    const int c0 = half * 16;
    // A layout (16x32 f16): lane L: M=L%16, base=(L/16)*8;
    // e<8 -> j=q*32+base+e ; e>=8 -> j=q*32+16+base+(e-8)   (all aligned v8h)
    const v8h* ap = (const v8h*)&xS[(c0 + p) * (UHP * UWP) + wid * UWP];
    v8f acc = {};
#pragma unroll
    for (int q = 0; q < KCH; ++q) {
      v8h alo = ap[q * 4 + hi];
      v8h ahi = ap[q * 4 + 2 + hi];
      v16h Af = __builtin_shufflevector(alo, ahi, 0, 1, 2, 3, 4, 5, 6, 7,
                                        8, 9, 10, 11, 12, 13, 14, 15);
      acc = __builtin_amdgcn_wmma_f32_16x16x32_f16(
          /*neg_a=*/false, Af, /*neg_b=*/false, Bf[q],
          /*c_mod=*/(short)0, acc, /*reuse_a=*/false, /*reuse_b=*/false);
    }
    // C/D f32 16x16 layout: VGPR r, lane L: N=L%16, M=r+8*(L/16)
#pragma unroll
    for (int r = 0; r < 8; ++r) {
      int ch = c0 + r + 8 * hi;
      out[((b * CIN + ch) * HH + h) * WW + w0 + p] = acc[r];
    }
  }
}

extern "C" void kernel_launch(void* const* d_in, const int* in_sizes, int n_in,
                              void* d_out, int out_size, void* d_ws, size_t ws_size,
                              hipStream_t stream) {
  (void)in_sizes; (void)n_in; (void)d_ws; (void)ws_size; (void)out_size;
  const float* x   = (const float*)d_in[0];   // input (2,32,256,256) f32
  const float* g   = (const float*)d_in[1];   // guide (2,3,256,256) f32
  const float* sig = (const float*)d_in[2];   // sigma scalar f32
  float* out = (float*)d_out;

  dim3 grid(WW / TPX, HH / TROWS, 2);   // 16 x 64 x 2 workgroups
  dim3 block(128);                      // 4 waves (wave32): one output row each
  bilateral_wmma_kernel<<<grid, block, 0, stream>>>(x, g, sig, out);
}
